// LocalSensitiveAttention_48670569398468
// MI455X (gfx1250) — matrix operand order repaired and checked by reading, not compiled, not run on hardware
//
#include <hip/hip_runtime.h>
#include <hip/hip_bf16.h>

// ---------------------------------------------------------------------------
// Location-sensitive attention, fused for MI455X (gfx1250, wave32, WMMA).
// Dominant cost: enc_proj GEMM [B*T,512]x[512,512] = 34.4 GFLOP -> bf16 WMMA.
// encoder_output (134MB) read exactly twice (energies pass + context pass).
// ---------------------------------------------------------------------------

#define B_DIM   64
#define T_DIM   1024
#define D_DEC   1024
#define DENC    512
#define A_DIM   512
#define NF_DIM  32
#define KCONV   31
#define MTILE   32          // t-rows per block in the fused kernel

typedef __attribute__((ext_vector_type(16))) __bf16 v16bf;
typedef __attribute__((ext_vector_type(8)))  __bf16 v8bf;
typedef __attribute__((ext_vector_type(8)))  float  v8f;

__device__ __forceinline__ unsigned short f2bfu(float f) {
    // round-to-nearest-even f32 -> bf16 (bit pattern)
    unsigned u = __builtin_bit_cast(unsigned, f);
    return (unsigned short)((u + 0x7FFFu + ((u >> 16) & 1u)) >> 16);
}

// --------------------------- prep kernels ----------------------------------

__global__ void cvt_bf16_kernel(const float* __restrict__ src,
                                unsigned short* __restrict__ dst, int n) {
    for (int i = blockIdx.x * blockDim.x + threadIdx.x; i < n;
         i += gridDim.x * blockDim.x)
        dst[i] = f2bfu(src[i]);
}

// mel_proj[b][a] = dot(mel[b,:], w_in[a,:]) + b_in[a]   (64x512 outputs)
__global__ void melproj_kernel(const float* __restrict__ mel,
                               const float* __restrict__ w_in,
                               const float* __restrict__ b_in,
                               float* __restrict__ mel_proj) {
    int idx = blockIdx.x * 256 + threadIdx.x;          // 32768 total
    int b = idx >> 9, a = idx & 511;
    const float4* mp = (const float4*)(mel + (size_t)b * D_DEC);
    const float4* wp = (const float4*)(w_in + (size_t)a * D_DEC);
    float acc = 0.f;
#pragma unroll 4
    for (int i = 0; i < D_DEC / 4; ++i) {
        float4 m = mp[i], w = wp[i];
        acc += m.x * w.x + m.y * w.y + m.z * w.z + m.w * w.w;
    }
    mel_proj[idx] = acc + b_in[a];
}

// ----------------------- fused energies kernel -----------------------------
// grid = B * (T/MTILE) = 2048 blocks, 256 threads = 8 waves.
// Per wave: m-subtile (16 rows) x 8 n-subtiles (128 cols of A) in f32 accs.
__global__ void __launch_bounds__(256)
energies_kernel(const float* __restrict__ enc,            // [B,T,DENC]
                const float* __restrict__ cum,            // [B,2,T]
                const unsigned char* __restrict__ mask,   // [B,T] (bool bytes)
                const float* __restrict__ conv_w,         // [NF,2,KCONV]
                const unsigned short* __restrict__ wenc_bf,  // [A,DENC] bf16
                const unsigned short* __restrict__ wloc_bf,  // [A,NF]   bf16
                const float* __restrict__ mel_proj,       // [B,A]
                const float* __restrict__ w_energy,       // [A]
                float* __restrict__ energies) {           // [B,T]
    __shared__ __align__(16) unsigned short enc_lds[MTILE * 520]; // pad 512+8
    __shared__ __align__(16) unsigned short conv_lds[MTILE * 40]; // pad 32+8
    __shared__ __align__(16) float convw_lds[NF_DIM * 2 * KCONV]; // 1984
    __shared__ float cum_lds[2 * 64];
    __shared__ float mel_lds[A_DIM];
    __shared__ float weng_lds[A_DIM];
    __shared__ float eparts[8][16];

    const int tid = threadIdx.x;
    const int b  = blockIdx.x >> 5;
    const int t0 = (blockIdx.x & 31) * MTILE;

    // ---- Phase A: global loads + f32->bf16 conversion into LDS ----
    {
        const float4* gp =
            (const float4*)(enc + (size_t)b * T_DIM * DENC + (size_t)t0 * DENC);
#pragma unroll
        for (int it = 0; it < (MTILE * DENC / 4) / 256; ++it) {
            int i = tid + it * 256;          // float4 index in the 32x512 tile
            int row = (i * 4) >> 9;
            int col = (i * 4) & 511;
            float4 v = gp[i];
            unsigned short* p = &enc_lds[row * 520 + col];
            // pack 4 bf16 as uint2 (8B aligned: col % 4 == 0, stride even)
            uint2 pk;
            pk.x = (unsigned)f2bfu(v.x) | ((unsigned)f2bfu(v.y) << 16);
            pk.y = (unsigned)f2bfu(v.z) | ((unsigned)f2bfu(v.w) << 16);
            *(uint2*)p = pk;
        }
    }
    // cumulative attention window: positions t0-15 .. t0+46 (62 valid)
    if (tid < 128) {
        int c = tid >> 6, i = tid & 63;
        float v = 0.f;
        int pos = t0 - 15 + i;
        if (i < MTILE + KCONV - 1 && pos >= 0 && pos < T_DIM)
            v = cum[(size_t)b * 2 * T_DIM + (size_t)c * T_DIM + pos];
        cum_lds[c * 64 + i] = v;
    }
    for (int i = tid; i < NF_DIM * 2 * KCONV; i += 256) convw_lds[i] = conv_w[i];
    for (int i = tid; i < A_DIM; i += 256) {
        mel_lds[i]  = mel_proj[b * A_DIM + i];
        weng_lds[i] = w_energy[i];
    }
    __syncthreads();

    // ---- Phase B: conv1d(2->32, k=31, SAME) for this t-tile ----
#pragma unroll
    for (int it = 0; it < (MTILE * NF_DIM) / 256; ++it) {
        int w  = tid + it * 256;
        int tl = w >> 5, f = w & 31;
        float acc = 0.f;
#pragma unroll
        for (int c = 0; c < 2; ++c) {
            const float* wf = &convw_lds[f * 2 * KCONV + c * KCONV];
            const float* xi = &cum_lds[c * 64 + tl];
#pragma unroll
            for (int k = 0; k < KCONV; ++k) acc += xi[k] * wf[k];
        }
        conv_lds[tl * 40 + f] = f2bfu(acc);
    }
    __syncthreads();

    // ---- Phase C: WMMA accumulation ----
    const int wave  = tid >> 5;
    const int lane  = tid & 31;
    const int mt    = wave & 1;            // m-subtile (rows t0+16*mt ..)
    const int nbase = (wave >> 1) * 8;     // 8 n-subtiles per wave
    const int lrow  = lane & 15;
    const int khalf = (lane >> 4) << 4;    // contiguous k half-select (matches B)
    const int kblo  = (lane < 16) ? 0 : 16;

    v8f acc[8];
    const v8f vzero = {0.f, 0.f, 0.f, 0.f, 0.f, 0.f, 0.f, 0.f};
#pragma unroll
    for (int j = 0; j < 8; ++j) acc[j] = vzero;

    // location term: single K=32 WMMA, A = conv features, B = w_loc (bf16)
    {
        const unsigned short* ap = &conv_lds[(mt * 16 + lrow) * 40 + khalf];
        v8bf lo = *(const v8bf*)ap;
        v8bf hi = *(const v8bf*)(ap + 8);
        v16bf a = __builtin_shufflevector(lo, hi, 0, 1, 2, 3, 4, 5, 6, 7,
                                          8, 9, 10, 11, 12, 13, 14, 15);
#pragma unroll
        for (int j = 0; j < 8; ++j) {
            int ncol = (nbase + j) * 16 + lrow;
            v16bf bf = *(const v16bf*)(wloc_bf + ncol * NF_DIM + kblo);
            acc[j] = __builtin_amdgcn_wmma_f32_16x16x32_bf16(
                false, a, false, bf, (short)0, acc[j], false, false);
        }
    }
    // enc_proj term: 16 K-steps of 32 over DENC=512
    for (int ks = 0; ks < DENC / 32; ++ks) {
        const unsigned short* ap =
            &enc_lds[(mt * 16 + lrow) * 520 + ks * 32 + khalf];
        v8bf lo = *(const v8bf*)ap;
        v8bf hi = *(const v8bf*)(ap + 8);
        v16bf a = __builtin_shufflevector(lo, hi, 0, 1, 2, 3, 4, 5, 6, 7,
                                          8, 9, 10, 11, 12, 13, 14, 15);
        int kb = ks * 32 + kblo;
#pragma unroll
        for (int j = 0; j < 8; ++j) {
            int ncol = (nbase + j) * 16 + lrow;
            v16bf bf = *(const v16bf*)(wenc_bf + ncol * DENC + kb);
            acc[j] = __builtin_amdgcn_wmma_f32_16x16x32_bf16(
                false, a, false, bf, (short)0, acc[j], false, false);
        }
    }

    // ---- Phase D: energy epilogue: sum_a w_e[a]*tanh(acc + mel_proj) ----
    float esum[8];
#pragma unroll
    for (int r = 0; r < 8; ++r) esum[r] = 0.f;
#pragma unroll
    for (int j = 0; j < 8; ++j) {
        int a0 = (nbase + j) * 16 + lrow;
        float mv = mel_lds[a0];
        float wv = weng_lds[a0];
#pragma unroll
        for (int r = 0; r < 8; ++r)
            esum[r] += wv * tanhf(acc[j][r] + mv);
    }
    // reduce across the 16 lanes sharing each output row (deterministic tree)
#pragma unroll
    for (int r = 0; r < 8; ++r) {
        float s = esum[r];
        s += __shfl_xor(s, 1, 32);
        s += __shfl_xor(s, 2, 32);
        s += __shfl_xor(s, 4, 32);
        s += __shfl_xor(s, 8, 32);
        if (lrow == 0) eparts[wave][r + (((lane >> 4)) << 3)] = s;
    }
    __syncthreads();
    if (tid < MTILE) {
        int mt2 = tid >> 4, rl = tid & 15;   // fixed-order cross-wave combine
        float e = eparts[mt2][rl] + eparts[mt2 + 2][rl] +
                  eparts[mt2 + 4][rl] + eparts[mt2 + 6][rl];
        int t = t0 + tid;
        energies[(size_t)b * T_DIM + t] =
            mask[(size_t)b * T_DIM + t] ? -__builtin_inff() : e;
    }
}

// --------------------------- softmax over T --------------------------------
__global__ void softmax_kernel(const float* __restrict__ energies,
                               float* __restrict__ attn) {
    __shared__ float red[256];
    __shared__ float row[T_DIM];
    int b = blockIdx.x, tid = threadIdx.x;
    float lmax = -__builtin_inff();
    for (int i = tid; i < T_DIM; i += 256) {
        float v = energies[(size_t)b * T_DIM + i];
        row[i] = v;
        lmax = fmaxf(lmax, v);
    }
    red[tid] = lmax;
    __syncthreads();
    for (int s = 128; s > 0; s >>= 1) {
        if (tid < s) red[tid] = fmaxf(red[tid], red[tid + s]);
        __syncthreads();
    }
    float mx = red[0];
    __syncthreads();
    float lsum = 0.f;
    for (int i = tid; i < T_DIM; i += 256) {
        float e = __expf(row[i] - mx);   // exp(-inf) == 0 handles the mask
        row[i] = e;
        lsum += e;
    }
    red[tid] = lsum;
    __syncthreads();
    for (int s = 128; s > 0; s >>= 1) {
        if (tid < s) red[tid] += red[tid + s];
        __syncthreads();
    }
    float inv = 1.0f / red[0];
    for (int i = tid; i < T_DIM; i += 256)
        attn[(size_t)b * T_DIM + i] = row[i] * inv;
}

// ------------------------ context = attn @ enc -----------------------------
// grid = B*4 blocks of 128 threads; each thread owns one e-column.
__global__ void context_kernel(const float* __restrict__ enc,
                               const float* __restrict__ attn,
                               float* __restrict__ ctx) {
    __shared__ float aw[T_DIM];
    int b = blockIdx.x >> 2;
    int e = ((blockIdx.x & 3) * 128) + threadIdx.x;
    for (int i = threadIdx.x; i < T_DIM; i += 128)
        aw[i] = attn[(size_t)b * T_DIM + i];
    __syncthreads();
    const float* ep = enc + (size_t)b * T_DIM * DENC + e;
    float acc = 0.f;
#pragma unroll 4
    for (int t = 0; t < T_DIM; ++t) acc += aw[t] * ep[(size_t)t * DENC];
    ctx[(size_t)b * DENC + e] = acc;
}

// ------------------------------- launcher ----------------------------------
extern "C" void kernel_launch(void* const* d_in, const int* in_sizes, int n_in,
                              void* d_out, int out_size, void* d_ws,
                              size_t ws_size, hipStream_t stream) {
    const float* mel_input = (const float*)d_in[0];
    const float* enc       = (const float*)d_in[1];
    const float* cum       = (const float*)d_in[2];
    const unsigned char* mask = (const unsigned char*)d_in[3]; // bool bytes
    const float* w_in      = (const float*)d_in[4];
    const float* b_in      = (const float*)d_in[5];
    const float* w_enc     = (const float*)d_in[6];
    const float* conv_w    = (const float*)d_in[7];
    const float* w_loc     = (const float*)d_in[8];
    const float* w_energy  = (const float*)d_in[9];

    // workspace layout (bytes)
    char* ws = (char*)d_ws;
    unsigned short* wenc_bf = (unsigned short*)(ws);                 // 512KB
    unsigned short* wloc_bf = (unsigned short*)(ws + 524288);        // 32KB
    float* mel_proj         = (float*)(ws + 557056);                 // 128KB
    float* energies         = (float*)(ws + 688128);                 // 256KB

    float* ctx_out  = (float*)d_out;                      // [B, DENC]
    float* attn_out = (float*)d_out + B_DIM * DENC;       // [B, T]

    cvt_bf16_kernel<<<256, 256, 0, stream>>>(w_enc, wenc_bf, A_DIM * DENC);
    cvt_bf16_kernel<<<64, 256, 0, stream>>>(w_loc, wloc_bf, A_DIM * NF_DIM);
    melproj_kernel<<<(B_DIM * A_DIM) / 256, 256, 0, stream>>>(
        mel_input, w_in, b_in, mel_proj);

    energies_kernel<<<B_DIM * (T_DIM / MTILE), 256, 0, stream>>>(
        enc, cum, mask, conv_w, wenc_bf, wloc_bf, mel_proj, w_energy, energies);

    softmax_kernel<<<B_DIM, 256, 0, stream>>>(energies, attn_out);
    context_kernel<<<B_DIM * 4, 128, 0, stream>>>(enc, attn_out, ctx_out);
}
